// EnergyForceModel_56831007260972
// MI455X (gfx1250) — matrix operand order repaired and checked by reading, not compiled, not run on hardware
//
#include <hip/hip_runtime.h>
#include <hip/hip_bf16.h>

typedef _Float16 half16 __attribute__((ext_vector_type(16)));
typedef _Float16 half8  __attribute__((ext_vector_type(8)));
typedef float    float8 __attribute__((ext_vector_type(8)));

#define NBASIS 7
#define NRAD   5
#define RCUT_F 6.0f
#define PI_F   3.14159265358979f

// ---------------------------------------------------------------------------
// Pair forward: geometry -> radial basis -> scatter-add moments m0..m3
// M layout per atom (200 f32): [0,5) m0 | [5,20) m1 | [20,65) m2 | [65,200) m3
// ---------------------------------------------------------------------------
__global__ void pair_fwd_kernel(const float* __restrict__ R, const int* __restrict__ Z,
                                const int* __restrict__ idx, const float* __restrict__ mu,
                                const float* __restrict__ Wrad, float* __restrict__ M, int P)
{
    int p = blockIdx.x * blockDim.x + threadIdx.x;
    if (p >= P) return;
    int i = idx[p], j = idx[P + p];
    float dx = R[3*j+0] - R[3*i+0];
    float dy = R[3*j+1] - R[3*i+1];
    float dz = R[3*j+2] - R[3*i+2];
    float dist = sqrtf(dx*dx + dy*dy + dz*dz + 1e-6f);
    float inv = 1.f / dist;
    float u[3] = {dx*inv, dy*inv, dz*inv};
    float fc = (dist < RCUT_F) ? 0.5f * (cosf(PI_F * dist / RCUT_F) + 1.f) : 0.f;
    float basis[NBASIS];
#pragma unroll
    for (int b = 0; b < NBASIS; ++b) {
        float t = dist - mu[b];
        basis[b] = expf(-t * t) * fc;           // ETA = 1
    }
    const float* wz = Wrad + Z[j] * (NRAD * NBASIS);
    float uu[9];
#pragma unroll
    for (int a = 0; a < 3; ++a)
#pragma unroll
        for (int b = 0; b < 3; ++b) uu[a*3+b] = u[a] * u[b];

    float* Mi = M + (size_t)i * 200;
#pragma unroll
    for (int r = 0; r < NRAD; ++r) {
        float rad = 0.f;
#pragma unroll
        for (int b = 0; b < NBASIS; ++b) rad += basis[b] * wz[r*NBASIS + b];
        atomicAdd(Mi + r, rad);
#pragma unroll
        for (int a = 0; a < 3; ++a) atomicAdd(Mi + 5 + r*3 + a, rad * u[a]);
#pragma unroll
        for (int k = 0; k < 9; ++k) atomicAdd(Mi + 20 + r*9 + k, rad * uu[k]);
#pragma unroll
        for (int k = 0; k < 9; ++k)
#pragma unroll
            for (int c = 0; c < 3; ++c)
                atomicAdd(Mi + 65 + r*27 + k*3 + c, rad * uu[k] * u[c]);
    }
}

// ---------------------------------------------------------------------------
// Invariant features: gm[n, 0..49] = [m0, triu(m1.m1), triu(m2.m2), triu(m3.m3)]
// Stored as f16 padded to 64 columns (WMMA A operand for layer 1).
// ---------------------------------------------------------------------------
__global__ void feat_kernel(const float* __restrict__ M, _Float16* __restrict__ gmH,
                            int N, int Npad)
{
    int n = blockIdx.x * blockDim.x + threadIdx.x;
    if (n >= Npad) return;
    _Float16* g = gmH + (size_t)n * 64;
    if (n >= N) {
        for (int t = 0; t < 64; ++t) g[t] = (_Float16)0.f;
        return;
    }
    const float* Mi = M + (size_t)n * 200;
    int t = 0;
    for (int r = 0; r < 5; ++r) g[t++] = (_Float16)Mi[r];
    for (int r = 0; r < 5; ++r)
        for (int s = r; s < 5; ++s) {
            float a = 0.f;
            for (int k = 0; k < 3; ++k) a += Mi[5 + r*3 + k] * Mi[5 + s*3 + k];
            g[t++] = (_Float16)a;
        }
    for (int r = 0; r < 5; ++r)
        for (int s = r; s < 5; ++s) {
            float a = 0.f;
            for (int k = 0; k < 9; ++k) a += Mi[20 + r*9 + k] * Mi[20 + s*9 + k];
            g[t++] = (_Float16)a;
        }
    for (int r = 0; r < 5; ++r)
        for (int s = r; s < 5; ++s) {
            float a = 0.f;
            for (int k = 0; k < 27; ++k) a += Mi[65 + r*27 + k] * Mi[65 + s*27 + k];
            g[t++] = (_Float16)a;
        }
    for (; t < 64; ++t) g[t] = (_Float16)0.f;
}

// ---------------------------------------------------------------------------
// Pack a f32 weight matrix (optionally transposed) into the WMMA B-fragment
// layout: for tile tn, k-block kb, lane l: 16 contiguous f16 with element e
// holding B[kb*32 + 16*(l>>4) + e][tn*16 + (l&15)].
// ---------------------------------------------------------------------------
__global__ void pack_b_kernel(const float* __restrict__ W, int Wr, int Wc,
                              int K, int tilesN, int transposed, _Float16* __restrict__ Bp)
{
    int id = blockIdx.x * blockDim.x + threadIdx.x;
    int KB = K >> 5;
    int total = tilesN * KB * 512;
    if (id >= total) return;
    int e    = id & 15;
    int lane = (id >> 4) & 31;
    int rem  = id >> 9;
    int kb   = rem % KB;
    int tn   = rem / KB;
    int k    = kb * 32 + 16 * (lane >> 4) + e;
    int col  = tn * 16 + (lane & 15);
    float v = 0.f;
    if (!transposed) { if (k < Wr && col < Wc) v = W[(size_t)k * Wc + col]; }
    else             { if (col < Wr && k < Wc) v = W[(size_t)col * Wc + k]; }
    Bp[id] = (_Float16)v;
}

// ---------------------------------------------------------------------------
// Epilogue helper for one 16x16 C tile held as v8f accumulator.
// ---------------------------------------------------------------------------
__device__ __forceinline__ void store_tile(float8 acc, int row0, int col, int mode,
                                           const float* __restrict__ bias,
                                           float* __restrict__ zout,
                                           _Float16* __restrict__ hout,
                                           const float* __restrict__ zref,
                                           float* __restrict__ fout, int ldc)
{
    union { float8 v; float e[8]; } cu; cu.v = acc;
    if (mode == 0) {
        float bc = bias[col];
#pragma unroll
        for (int v = 0; v < 8; ++v) {
            int row = row0 + v;
            float z = cu.e[v] + bc;
            zout[(size_t)row * ldc + col] = z;
            float s = 1.f / (1.f + expf(-z));
            hout[(size_t)row * ldc + col] = (_Float16)(z * s);
        }
    } else if (mode == 1) {
#pragma unroll
        for (int v = 0; v < 8; ++v) {
            int row = row0 + v;
            float z = zref[(size_t)row * ldc + col];
            float s = 1.f / (1.f + expf(-z));
            hout[(size_t)row * ldc + col] = (_Float16)(cu.e[v] * (s * (1.f + z * (1.f - s))));
        }
    } else {
#pragma unroll
        for (int v = 0; v < 8; ++v) {
            int row = row0 + v;
            fout[(size_t)row * ldc + col] = cu.e[v];
        }
    }
}

// ---------------------------------------------------------------------------
// Register-blocked WMMA GEMM: one wave computes a 32x32 macro-tile (2x2 of
// 16x16 WMMA tiles), so each A/B fragment load feeds two v_wmma ops
// (2x arithmetic intensity vs 1 tile/wave; halves L2 traffic).
// mode 0: z = acc + bias; store z (f32) and swish(z) (f16)
// mode 1: store acc * swish'(zref) as f16
// mode 2: store acc as f32
// ---------------------------------------------------------------------------
__global__ void __launch_bounds__(256)
wmma_gemm_kernel(const _Float16* __restrict__ A, const _Float16* __restrict__ Bp,
                 int K, int tilesN2, int totalTiles2, int mode,
                 const float* __restrict__ bias,
                 float* __restrict__ zout, _Float16* __restrict__ hout,
                 const float* __restrict__ zref, float* __restrict__ fout, int ldc)
{
    int wave = (int)((blockIdx.x * blockDim.x + threadIdx.x) >> 5);
    if (wave >= totalTiles2) return;           // wave-uniform exit (EXEC stays full)
    int lane = threadIdx.x & 31;
    int tm2 = wave / tilesN2, tn2 = wave % tilesN2;
    int half = lane >> 4;
    int m    = lane & 15;
    int KB   = K >> 5;
    const _Float16* arow0 = A + (size_t)((tm2*2 + 0) * 16 + m) * K;
    const _Float16* arow1 = A + (size_t)((tm2*2 + 1) * 16 + m) * K;
    const _Float16* bb0   = Bp + (size_t)(tn2*2 + 0) * KB * 512;
    const _Float16* bb1   = Bp + (size_t)(tn2*2 + 1) * KB * 512;
    float8 acc00 = {0.f,0.f,0.f,0.f,0.f,0.f,0.f,0.f};
    float8 acc01 = {0.f,0.f,0.f,0.f,0.f,0.f,0.f,0.f};
    float8 acc10 = {0.f,0.f,0.f,0.f,0.f,0.f,0.f,0.f};
    float8 acc11 = {0.f,0.f,0.f,0.f,0.f,0.f,0.f,0.f};
    for (int kb = 0; kb < KB; ++kb) {
        int aoff = kb * 32 + 16 * half;
        union { half16 v; half8 h[2]; } a0, a1;
        a0.h[0] = *(const half8*)(arow0 + aoff);        // K = 16*half..+7
        a0.h[1] = *(const half8*)(arow0 + aoff + 8);    // K = 16*half+8..+15
        a1.h[0] = *(const half8*)(arow1 + aoff);
        a1.h[1] = *(const half8*)(arow1 + aoff + 8);
        size_t boff = ((size_t)kb * 32 + lane) * 16;
        half16 b0 = *(const half16*)(bb0 + boff);
        half16 b1 = *(const half16*)(bb1 + boff);
        acc00 = __builtin_amdgcn_wmma_f32_16x16x32_f16(false, a0.v, false, b0,
                                                       (short)0, acc00, false, false);
        acc01 = __builtin_amdgcn_wmma_f32_16x16x32_f16(false, a0.v, false, b1,
                                                       (short)0, acc01, false, false);
        acc10 = __builtin_amdgcn_wmma_f32_16x16x32_f16(false, a1.v, false, b0,
                                                       (short)0, acc10, false, false);
        acc11 = __builtin_amdgcn_wmma_f32_16x16x32_f16(false, a1.v, false, b1,
                                                       (short)0, acc11, false, false);
    }
    int col0 = (tn2*2 + 0) * 16 + m;
    int col1 = (tn2*2 + 1) * 16 + m;
    int row0 = (tm2*2 + 0) * 16 + 8 * half;
    int row1 = (tm2*2 + 1) * 16 + 8 * half;
    store_tile(acc00, row0, col0, mode, bias, zout, hout, zref, fout, ldc);
    store_tile(acc01, row0, col1, mode, bias, zout, hout, zref, fout, ldc);
    store_tile(acc10, row1, col0, mode, bias, zout, hout, zref, fout, ldc);
    store_tile(acc11, row1, col1, mode, bias, zout, hout, zref, fout, ldc);
}

// ---------------------------------------------------------------------------
// Head: e3 = h2 . W3 + b3 ; e = e3*scale[Z]+shift[Z], masked Z>0.
// Accumulate energy; stash de[n] = dE/de3 = mask*scale[Z].
// ---------------------------------------------------------------------------
__global__ void head_kernel(const _Float16* __restrict__ h2H, const float* __restrict__ W3,
                            const float* __restrict__ b3, const int* __restrict__ Z,
                            const float* __restrict__ scale, const float* __restrict__ shift,
                            float* __restrict__ de, float* __restrict__ energy, int N)
{
    int n = blockIdx.x * blockDim.x + threadIdx.x;
    if (n >= N) return;
    const _Float16* h = h2H + (size_t)n * 512;
    float acc = 0.f;
    for (int k = 0; k < 512; ++k) acc += (float)h[k] * W3[k];
    int z = Z[n];
    float e = (acc + b3[0]) * scale[z] + shift[z];
    float m = (z > 0) ? 1.f : 0.f;
    de[n] = m * scale[z];
    atomicAdd(energy, m * e);
}

// dz2 = de * W3 * swish'(z2)  (f16, WMMA A operand for the W2^T GEMM)
__global__ void bw3_kernel(const float* __restrict__ de, const float* __restrict__ W3,
                           const float* __restrict__ z2, _Float16* __restrict__ dz2H,
                           int N, int Npad)
{
    int id = blockIdx.x * blockDim.x + threadIdx.x;
    if (id >= Npad * 512) return;
    int n = id >> 9, k = id & 511;
    if (n >= N) { dz2H[id] = (_Float16)0.f; return; }
    float z = z2[id];
    float s = 1.f / (1.f + expf(-z));
    dz2H[id] = (_Float16)(de[n] * W3[k] * (s * (1.f + z * (1.f - s))));
}

// ---------------------------------------------------------------------------
// Moment gradients: dM = G_k . m_k with symmetric G built from triu feature
// grads (diagonal doubled for quadratic contractions).
// ---------------------------------------------------------------------------
__global__ void dmom_kernel(const float* __restrict__ ggm, const float* __restrict__ M,
                            float* __restrict__ dM, int N)
{
    int n = blockIdx.x * blockDim.x + threadIdx.x;
    if (n >= N) return;
    const float* g  = ggm + (size_t)n * 64;
    const float* Mi = M   + (size_t)n * 200;
    float* Di       = dM  + (size_t)n * 200;
    float G1[25], G2[25], G3[25];
    int t = 5;
    for (int r = 0; r < 5; ++r)
        for (int s = r; s < 5; ++s) { float v = g[t++];
            if (r == s) G1[r*5+r] = 2.f*v; else { G1[r*5+s] = v; G1[s*5+r] = v; } }
    for (int r = 0; r < 5; ++r)
        for (int s = r; s < 5; ++s) { float v = g[t++];
            if (r == s) G2[r*5+r] = 2.f*v; else { G2[r*5+s] = v; G2[s*5+r] = v; } }
    for (int r = 0; r < 5; ++r)
        for (int s = r; s < 5; ++s) { float v = g[t++];
            if (r == s) G3[r*5+r] = 2.f*v; else { G3[r*5+s] = v; G3[s*5+r] = v; } }
    for (int r = 0; r < 5; ++r) Di[r] = g[r];
    for (int r = 0; r < 5; ++r)
        for (int k = 0; k < 3; ++k) {
            float a = 0.f;
            for (int s = 0; s < 5; ++s) a += G1[r*5+s] * Mi[5 + s*3 + k];
            Di[5 + r*3 + k] = a;
        }
    for (int r = 0; r < 5; ++r)
        for (int k = 0; k < 9; ++k) {
            float a = 0.f;
            for (int s = 0; s < 5; ++s) a += G2[r*5+s] * Mi[20 + s*9 + k];
            Di[20 + r*9 + k] = a;
        }
    for (int r = 0; r < 5; ++r)
        for (int k = 0; k < 27; ++k) {
            float a = 0.f;
            for (int s = 0; s < 5; ++s) a += G3[r*5+s] * Mi[65 + s*27 + k];
            Di[65 + r*27 + k] = a;
        }
}

// ---------------------------------------------------------------------------
// Pair backward: recompute geometry/radial, chain dM -> (radial, u, dist) -> dr
// and atomically scatter position gradients.
// ---------------------------------------------------------------------------
__global__ void pair_bwd_kernel(const float* __restrict__ R, const int* __restrict__ Z,
                                const int* __restrict__ idx, const float* __restrict__ mu,
                                const float* __restrict__ Wrad, const float* __restrict__ dM,
                                float* __restrict__ gradR, int P)
{
    int p = blockIdx.x * blockDim.x + threadIdx.x;
    if (p >= P) return;
    int i = idx[p], j = idx[P + p];
    float dx = R[3*j+0] - R[3*i+0];
    float dy = R[3*j+1] - R[3*i+1];
    float dz = R[3*j+2] - R[3*i+2];
    float dist = sqrtf(dx*dx + dy*dy + dz*dz + 1e-6f);
    float inv = 1.f / dist;
    float u[3] = {dx*inv, dy*inv, dz*inv};
    float fc = (dist < RCUT_F) ? 0.5f * (cosf(PI_F * dist / RCUT_F) + 1.f) : 0.f;
    float basis[NBASIS];
#pragma unroll
    for (int b = 0; b < NBASIS; ++b) {
        float t = dist - mu[b];
        basis[b] = expf(-t * t) * fc;
    }
    const float* wz = Wrad + Z[j] * (NRAD * NBASIS);
    float radial[NRAD];
#pragma unroll
    for (int r = 0; r < NRAD; ++r) {
        float a = 0.f;
#pragma unroll
        for (int b = 0; b < NBASIS; ++b) a += basis[b] * wz[r*NBASIS + b];
        radial[r] = a;
    }
    float uu[9];
#pragma unroll
    for (int a = 0; a < 3; ++a)
#pragma unroll
        for (int b = 0; b < 3; ++b) uu[a*3+b] = u[a] * u[b];

    const float* Di = dM + (size_t)i * 200;
    float gr[NRAD];
    float gu[3] = {0.f, 0.f, 0.f};
    for (int r = 0; r < NRAD; ++r) {
        const float* d1 = Di + 5  + r*3;
        const float* d2 = Di + 20 + r*9;
        const float* d3 = Di + 65 + r*27;
        float s = Di[r];
        for (int a = 0; a < 3; ++a) s += d1[a] * u[a];
        for (int k = 0; k < 9; ++k) s += d2[k] * uu[k];
        for (int k = 0; k < 9; ++k)
            for (int c = 0; c < 3; ++c) s += d3[k*3 + c] * uu[k] * u[c];
        gr[r] = s;
        float rad = radial[r];
        for (int x = 0; x < 3; ++x) {
            float t = d1[x];
            for (int b = 0; b < 3; ++b) t += (d2[x*3 + b] + d2[b*3 + x]) * u[b];
            for (int a = 0; a < 3; ++a)
                for (int b = 0; b < 3; ++b) {
                    float w = uu[a*3 + b];
                    t += d3[x*9 + a*3 + b] * w;
                    t += d3[a*9 + x*3 + b] * w;
                    t += d3[a*9 + b*3 + x] * w;
                }
            gu[x] += rad * t;
        }
    }
    // dL/dbasis -> dL/ddist
    float gb[NBASIS];
#pragma unroll
    for (int b = 0; b < NBASIS; ++b) {
        float a = 0.f;
#pragma unroll
        for (int r = 0; r < NRAD; ++r) a += gr[r] * wz[r*NBASIS + b];
        gb[b] = a;
    }
    float dfc = (dist < RCUT_F) ? (-0.5f * PI_F / RCUT_F) * sinf(PI_F * dist / RCUT_F) : 0.f;
    float gd = 0.f;
#pragma unroll
    for (int b = 0; b < NBASIS; ++b) {
        float t  = dist - mu[b];
        float ex = expf(-t * t);
        gd += gb[b] * (ex * (-2.f * t) * fc + ex * dfc);
    }
    float gdotu = gu[0]*u[0] + gu[1]*u[1] + gu[2]*u[2];
#pragma unroll
    for (int x = 0; x < 3; ++x) {
        float gx = gd * u[x] + (gu[x] - gdotu * u[x]) * inv;   // dL/d(dr_x)
        atomicAdd(gradR + 3*j + x,  gx);
        atomicAdd(gradR + 3*i + x, -gx);
    }
}

__global__ void forces_kernel(const float* __restrict__ gradR, float* __restrict__ out, int n3)
{
    int id = blockIdx.x * blockDim.x + threadIdx.x;
    if (id < n3) out[1 + id] = -gradR[id];
}

// ---------------------------------------------------------------------------
extern "C" void kernel_launch(void* const* d_in, const int* in_sizes, int n_in,
                              void* d_out, int out_size, void* d_ws, size_t ws_size,
                              hipStream_t stream)
{
    const float* R     = (const float*)d_in[0];
    const int*   Z     = (const int*)  d_in[1];
    const int*   idx   = (const int*)  d_in[2];
    const float* mu    = (const float*)d_in[4];
    const float* Wrad  = (const float*)d_in[5];
    const float* W1    = (const float*)d_in[6];
    const float* b1    = (const float*)d_in[7];
    const float* W2    = (const float*)d_in[8];
    const float* b2    = (const float*)d_in[9];
    const float* W3    = (const float*)d_in[10];
    const float* b3    = (const float*)d_in[11];
    const float* scale = (const float*)d_in[12];
    const float* shift = (const float*)d_in[13];
    float* out = (float*)d_out;

    const int N    = in_sizes[0] / 3;
    const int P    = in_sizes[2] / 2;
    const int Npad = ((N + 31) / 32) * 32;     // rows padded for 2x2 macro-tiles
    const int tilesM2 = Npad / 32;
    const int H = 512;

    char* ws = (char*)d_ws;
    size_t off = 0;
    auto alloc = [&](size_t bytes) -> void* {
        void* p = ws + off;
        off = (off + bytes + 255) & ~(size_t)255;
        return p;
    };
    float*     M     = (float*)    alloc((size_t)N    * 200 * 4);
    float*     dMb   = (float*)    alloc((size_t)N    * 200 * 4);
    _Float16*  gmH   = (_Float16*) alloc((size_t)Npad * 64  * 2);
    float*     z1    = (float*)    alloc((size_t)Npad * H   * 4);
    _Float16*  h1H   = (_Float16*) alloc((size_t)Npad * H   * 2);
    float*     z2    = (float*)    alloc((size_t)Npad * H   * 4);
    _Float16*  h2H   = (_Float16*) alloc((size_t)Npad * H   * 2);
    _Float16*  dz2H  = (_Float16*) alloc((size_t)Npad * H   * 2);
    _Float16*  dz1H  = (_Float16*) alloc((size_t)Npad * H   * 2);
    float*     ggm   = (float*)    alloc((size_t)Npad * 64  * 4);
    float*     de    = (float*)    alloc((size_t)N    * 4);
    float*     gradR = (float*)    alloc((size_t)N    * 3 * 4);
    _Float16*  W1p   = (_Float16*) alloc((size_t)64  * H * 2);
    _Float16*  W2p   = (_Float16*) alloc((size_t)H   * H * 2);
    _Float16*  W2Tp  = (_Float16*) alloc((size_t)H   * H * 2);
    _Float16*  W1Tp  = (_Float16*) alloc((size_t)H   * 64 * 2);

    hipMemsetAsync(M,     0, (size_t)N * 200 * 4, stream);
    hipMemsetAsync(gradR, 0, (size_t)N * 3 * 4, stream);
    hipMemsetAsync(d_out, 0, (size_t)out_size * 4, stream);

    // Weight packing into WMMA B-fragment order (f16).
    {
        int tot;
        tot = 32 * (64  >> 5) * 512;
        pack_b_kernel<<<(tot + 255) / 256, 256, 0, stream>>>(W1, 50, H, 64, 32, 0, W1p);
        tot = 32 * (H   >> 5) * 512;
        pack_b_kernel<<<(tot + 255) / 256, 256, 0, stream>>>(W2, H, H, H, 32, 0, W2p);
        pack_b_kernel<<<(tot + 255) / 256, 256, 0, stream>>>(W2, H, H, H, 32, 1, W2Tp);
        tot = 4  * (H   >> 5) * 512;
        pack_b_kernel<<<(tot + 255) / 256, 256, 0, stream>>>(W1, 50, H, H, 4, 1, W1Tp);
    }

    pair_fwd_kernel<<<(P + 255) / 256, 256, 0, stream>>>(R, Z, idx, mu, Wrad, M, P);
    feat_kernel<<<(Npad + 255) / 256, 256, 0, stream>>>(M, gmH, N, Npad);

    // Layer 1: z1 = gm @ W1 + b1 ; h1 = swish(z1)
    {
        int tiles2 = tilesM2 * 16;             // tilesN2 = 32/2
        wmma_gemm_kernel<<<(tiles2 + 7) / 8, 256, 0, stream>>>(
            gmH, W1p, 64, 16, tiles2, 0, b1, z1, h1H, nullptr, nullptr, H);
    }
    // Layer 2: z2 = h1 @ W2 + b2 ; h2 = swish(z2)
    {
        int tiles2 = tilesM2 * 16;
        wmma_gemm_kernel<<<(tiles2 + 7) / 8, 256, 0, stream>>>(
            h1H, W2p, H, 16, tiles2, 0, b2, z2, h2H, nullptr, nullptr, H);
    }
    head_kernel<<<(N + 255) / 256, 256, 0, stream>>>(h2H, W3, b3, Z, scale, shift,
                                                     de, out, N);
    bw3_kernel<<<((Npad * H) + 255) / 256, 256, 0, stream>>>(de, W3, z2, dz2H, N, Npad);
    // dh1 = dz2 @ W2^T ; dz1 = dh1 * swish'(z1)
    {
        int tiles2 = tilesM2 * 16;
        wmma_gemm_kernel<<<(tiles2 + 7) / 8, 256, 0, stream>>>(
            dz2H, W2Tp, H, 16, tiles2, 1, nullptr, nullptr, dz1H, z1, nullptr, H);
    }
    // dgm = dz1 @ W1^T (cols >= 50 are zero via padded W1Tp)
    {
        int tiles2 = tilesM2 * 2;              // tilesN2 = 4/2
        wmma_gemm_kernel<<<(tiles2 + 7) / 8, 256, 0, stream>>>(
            dz1H, W1Tp, H, 2, tiles2, 2, nullptr, nullptr, nullptr, nullptr, ggm, 64);
    }
    dmom_kernel<<<(N + 255) / 256, 256, 0, stream>>>(ggm, M, dMb, N);
    pair_bwd_kernel<<<(P + 255) / 256, 256, 0, stream>>>(R, Z, idx, mu, Wrad, dMb, gradR, P);
    forces_kernel<<<((N * 3) + 255) / 256, 256, 0, stream>>>(gradR, out, N * 3);
}
